// CapsNet_26044681683575
// MI455X (gfx1250) — compile-verified, matrix operand order checked by reading, not run on hardware
//
#include <hip/hip_runtime.h>

typedef __attribute__((ext_vector_type(16))) _Float16 v16h;
typedef __attribute__((ext_vector_type(8)))  float    v8f;

// ---------------- sizes ----------------
#define BATCH 8
#define IMG 72
#define FEAT 64
#define C1 256
#define PC1_N 512          // 32*16
#define PC1_K 6400         // 256*25
#define PC1_M 7200         // 8*30*30
#define R1 30
#define R2 28
#define NCAPS 32
#define CAPD 8
#define NCLS 10
#define DIGD 16
#define NPRIM 25088        // 28*28*32

// ---------------- workspace offsets (bytes) ----------------
static const size_t H_OFF    = 0;            // conv1 out [8,256,64,64] f32 (33,554,432 B)
static const size_t PC2O_OFF = 0;            // pc2 out NHWC [8,30,30,256] f32 (reuses h)
static const size_t PC1O_OFF = 7372800;      // pc1 out NHWC [7200,512] f32 (reuses h tail)
static const size_t A_OFF    = 33554432;     // im2col f16 [7200,6400]
static const size_t WT_OFF   = 125714432;    // pc1 weights f16, swizzled [6400/32][512][32]
static const size_t UHAT_OFF = 0;            // u_hat [8,25088,10,16] f32 (reuses everything above)
static const size_t U_OFF    = 128450560;    // u [8,25088,8] f32
static const size_t BLOG_OFF = 134873088;    // b_logits [8,25088,10] f32
static const size_t S_OFF    = 142901248;    // s [8,10,16]
static const size_t V_OFF    = 142906368;    // v [8,10,16]
static const size_t VM_OFF   = 142911488;    // v_masked [8,16]
static const size_t H1_OFF   = 142912000;    // fc1 out [8,512]
static const size_t H2_OFF   = 142928384;    // fc2 out [8,1024]
static const size_t RECH_OFF = 142961152;    // fc3 out [8,1296]

// ---------------- misc ----------------
__global__ void zero_kernel(float* __restrict__ p, long n) {
  long i = (long)blockIdx.x * blockDim.x + threadIdx.x;
  if (i < n) p[i] = 0.f;
}

// ---------------- conv1: 9x9, 1->256, stride1, valid, relu ----------------
__global__ void conv1_kernel(const float* __restrict__ x, const float* __restrict__ w,
                             const float* __restrict__ bias, float* __restrict__ out) {
  int idx = blockIdx.x * blockDim.x + threadIdx.x;
  const int total = BATCH * C1 * FEAT * FEAT;
  if (idx >= total) return;
  int ox = idx & 63; int t = idx >> 6;
  int oy = t & 63;   t >>= 6;
  int oc = t & 255;  int b = t >> 8;
  const float* xp = x + ((size_t)b * IMG + oy) * IMG + ox;
  const float* wp = w + (size_t)oc * 81;
  float acc = bias[oc];
  #pragma unroll
  for (int ky = 0; ky < 9; ky++)
    #pragma unroll
    for (int kx = 0; kx < 9; kx++)
      acc += xp[ky * IMG + kx] * wp[ky * 9 + kx];
  out[idx] = fmaxf(acc, 0.f);
}

// ---------------- im2col for pc1 (f32 -> f16) ----------------
__global__ void im2col_pc1(const float* __restrict__ h, _Float16* __restrict__ A) {
  size_t idx = (size_t)blockIdx.x * blockDim.x + threadIdx.x;
  const size_t total = (size_t)PC1_M * PC1_K;
  if (idx >= total) return;
  int k = (int)(idx % PC1_K);
  size_t m = idx / PC1_K;
  int ox = (int)(m % R1); size_t t = m / R1;
  int oy = (int)(t % R1); int b = (int)(t / R1);
  int kx = k % 5; int t2 = k / 5;
  int ky = t2 % 5; int c = t2 / 5;
  int iy = oy * 2 + ky, ix = ox * 2 + kx;
  A[idx] = (_Float16)h[(((size_t)b * C1 + c) * FEAT + iy) * FEAT + ix];
}

// ------- pc1 weight: f32 [512,6400] -> f16 swizzled [K/32][512][32] (fragment-ready) -----
__global__ void wt_pc1(const float* __restrict__ w, _Float16* __restrict__ Wt) {
  size_t idx = (size_t)blockIdx.x * blockDim.x + threadIdx.x;
  const size_t total = (size_t)PC1_K * PC1_N;
  if (idx >= total) return;
  int kk = (int)(idx & 31); size_t t = idx >> 5;
  int n  = (int)(t % PC1_N);
  int kb = (int)(t / PC1_N);
  Wt[idx] = (_Float16)w[(size_t)n * PC1_K + kb * 32 + kk];
}

// ---------------- WMMA GEMM: C[M,N] = A[M,K] * Bsw + bias[N] (f16 in, f32 out) ----------
// Bsw is pre-swizzled [K/32][N][32].
// block: 256 threads = 8 waves; block tile 128x128; wave tile 32x64 (2x4 WMMA frags)
#define BM 128
#define BN 128
#define BK 32
#define TSTR 40            // LDS row stride in halves (32 + 8 pad)
__device__ __forceinline__ v16h load_frag(const _Float16* __restrict__ p, int kbase) {
  v16h f;
  #pragma unroll
  for (int e = 0; e < 8; e++) f[e] = p[kbase + e];         // VGPR0..3: K = kbase..kbase+7
  #pragma unroll
  for (int e = 0; e < 8; e++) f[8 + e] = p[16 + kbase + e]; // VGPR4..7: K = 16+kbase..
  return f;
}

__global__ __launch_bounds__(256) void wmma_gemm_bias(
    const _Float16* __restrict__ A, const _Float16* __restrict__ Bsw,
    const float* __restrict__ bias, float* __restrict__ C,
    int M, int N, int K) {
  __shared__ _Float16 As[BM * TSTR];   // [row][kk]
  __shared__ _Float16 Bs[BN * TSTR];   // [col][kk]
  int tid = threadIdx.x;
  int lane = tid & 31, wave = tid >> 5;
  int wm0 = (wave & 3) * 32;           // 4 waves along M (32 rows each)
  int wn0 = (wave >> 2) * 64;          // 2 waves along N (64 cols each)
  int m0 = blockIdx.x * BM;
  int n0 = blockIdx.y * BN;

  int arow = tid >> 1;                 // 0..127
  int aseg = (tid & 1) * 16;           // halves
  int mrow = lane & 15;
  int kbase = (lane >> 4) * 8;         // lane half selects K 0-7 / 8-15 (+16 in upper VGPRs)

  v8f acc[2][4] = {};

  for (int k0 = 0; k0 < K; k0 += BK) {
    __syncthreads();
    {   // stage A tile [BM x BK]
      int gm = m0 + arow;
      uint4 val = make_uint4(0u, 0u, 0u, 0u);
      if (gm < M) val = *(const uint4*)(A + (size_t)gm * K + k0 + aseg);
      *(uint4*)(&As[arow * TSTR + aseg]) = val;
      if (k0 + BK < K && gm < M)
        __builtin_prefetch((const void*)(A + (size_t)gm * K + k0 + BK + aseg), 0, 0);
    }
    {   // stage B tile [BN cols x BK] from swizzled [kb][n][32] -- straight copy
      const _Float16* bsrc = Bsw + (((size_t)(k0 >> 5) * N) + n0 + arow) * 32 + aseg;
      uint4 val = *(const uint4*)bsrc;
      *(uint4*)(&Bs[arow * TSTR + aseg]) = val;
      if (k0 + BK < K)
        __builtin_prefetch((const void*)(bsrc + (size_t)N * 32), 0, 0);
    }
    __syncthreads();

    v16h afr[2], bfr[4];
    #pragma unroll
    for (int i = 0; i < 2; i++)
      afr[i] = load_frag(&As[(wm0 + i * 16 + mrow) * TSTR], kbase);
    #pragma unroll
    for (int j = 0; j < 4; j++)
      bfr[j] = load_frag(&Bs[(wn0 + j * 16 + mrow) * TSTR], kbase);

    #pragma unroll
    for (int i = 0; i < 2; i++)
      #pragma unroll
      for (int j = 0; j < 4; j++)
        acc[i][j] = __builtin_amdgcn_wmma_f32_16x16x32_f16(
            false, afr[i], false, bfr[j], (short)0, acc[i][j], false, false);
  }

  int rowoff = (lane >> 4) * 8;        // D layout: VGPR r -> M=r (lanes 0-15) / M=8+r
  #pragma unroll
  for (int i = 0; i < 2; i++)
    #pragma unroll
    for (int j = 0; j < 4; j++) {
      int cc = n0 + wn0 + j * 16 + (lane & 15);
      float bv = bias ? bias[cc] : 0.f;
      #pragma unroll
      for (int r = 0; r < 8; r++) {
        int rr = m0 + wm0 + i * 16 + rowoff + r;
        if (rr < M) C[(size_t)rr * N + cc] = acc[i][j][r] + bv;
      }
    }
}

// ---------------- pc2: grouped 3x3 pad1, 32 groups, 16->8, NHWC ----------------
__global__ void pc2_kernel(const float* __restrict__ in, const float* __restrict__ w,
                           const float* __restrict__ bias, float* __restrict__ out) {
  int idx = blockIdx.x * blockDim.x + threadIdx.x;
  const int total = BATCH * R1 * R1 * 256;
  if (idx >= total) return;
  int c = idx & 255; int t = idx >> 8;
  int x = t % R1; t /= R1;
  int y = t % R1; int b = t / R1;
  int g = c >> 3;
  float acc = bias[c];
  for (int ky = 0; ky < 3; ky++) {
    int iy = y + ky - 1;
    if (iy < 0 || iy >= R1) continue;
    for (int kx = 0; kx < 3; kx++) {
      int ix = x + kx - 1;
      if (ix < 0 || ix >= R1) continue;
      const float* ip = in + (((size_t)b * 900 + iy * R1 + ix) * PC1_N) + g * 16;
      const float* wp = w + (size_t)c * 16 * 9 + ky * 3 + kx;
      #pragma unroll
      for (int ic = 0; ic < 16; ic++) acc += ip[ic] * wp[ic * 9];
    }
  }
  out[idx] = acc;
}

// ---------------- pc3: grouped 3x3 valid, 32 groups, 8->8; writes capsule layout ------
__global__ void pc3_kernel(const float* __restrict__ in, const float* __restrict__ w,
                           const float* __restrict__ bias, float* __restrict__ u) {
  int idx = blockIdx.x * blockDim.x + threadIdx.x;
  const int total = BATCH * R2 * R2 * 256;
  if (idx >= total) return;
  int c = idx & 255; int t = idx >> 8;
  int x = t % R2; t /= R2;
  int y = t % R2; int b = t / R2;
  int g = c >> 3;
  float acc = bias[c];
  #pragma unroll
  for (int ky = 0; ky < 3; ky++)
    #pragma unroll
    for (int kx = 0; kx < 3; kx++) {
      const float* ip = in + (((size_t)b * 900 + (y + ky) * R1 + (x + kx)) * 256) + g * 8;
      const float* wp = w + (size_t)c * 8 * 9 + ky * 3 + kx;
      #pragma unroll
      for (int ic = 0; ic < 8; ic++) acc += ip[ic] * wp[ic * 9];
    }
  // u layout [b][s=y*28+x][cap][dim] == NHWC with channel c = cap*8+dim
  u[((size_t)b * 784 + y * R2 + x) * 256 + c] = acc;
}

// ---------------- squash over last 8 dims, in place ----------------
__global__ void squash_u_kernel(float* __restrict__ u) {
  int idx = blockIdx.x * blockDim.x + threadIdx.x;
  const int total = BATCH * NPRIM;
  if (idx >= total) return;
  float* p = u + (size_t)idx * CAPD;
  float sq = 0.f;
  #pragma unroll
  for (int i = 0; i < CAPD; i++) sq += p[i] * p[i];
  float scale = sq / (sqrtf(sq) * (1.f + sq));
  #pragma unroll
  for (int i = 0; i < CAPD; i++) p[i] *= scale;
}

// ---------------- u_hat[b,n,k,j] = sum_i u[b,n,i] * W[n,k,i,j] ----------------
__global__ void uhat_kernel(const float* __restrict__ u, const float* __restrict__ W,
                            float* __restrict__ uhat) {
  long idx = (long)blockIdx.x * blockDim.x + threadIdx.x;
  const long total = (long)BATCH * NPRIM * NCLS * DIGD;
  if (idx >= total) return;
  int j = (int)(idx & 15); long t = idx >> 4;
  int k = (int)(t % NCLS); t /= NCLS;
  int n = (int)(t % NPRIM); int b = (int)(t / NPRIM);
  const float* up = u + ((size_t)b * NPRIM + n) * CAPD;
  const float* wp = W + (((size_t)n * NCLS + k) * CAPD) * DIGD + j;
  float acc = 0.f;
  #pragma unroll
  for (int i = 0; i < CAPD; i++) acc += up[i] * wp[i * DIGD];
  uhat[idx] = acc;
}

// ---------------- routing: s[b,k,j] += softmax_k(blog[b,n,:]) * uhat ----------------
__global__ __launch_bounds__(256) void routing_reduce(const float* __restrict__ uhat,
                                                      const float* __restrict__ blog,
                                                      float* __restrict__ s) {
  __shared__ float ls[NCLS * DIGD];
  int b = blockIdx.y;
  for (int i = threadIdx.x; i < NCLS * DIGD; i += 256) ls[i] = 0.f;
  __syncthreads();
  int nbeg = blockIdx.x * 784;   // 25088 / 32 chunks
  for (int n = nbeg + threadIdx.x; n < nbeg + 784; n += 256) {
    const float* lg = blog + ((size_t)b * NPRIM + n) * NCLS;
    float mx = lg[0];
    #pragma unroll
    for (int k = 1; k < NCLS; k++) mx = fmaxf(mx, lg[k]);
    float e[NCLS]; float sum = 0.f;
    #pragma unroll
    for (int k = 0; k < NCLS; k++) { e[k] = __expf(lg[k] - mx); sum += e[k]; }
    float inv = 1.f / sum;
    const float* up = uhat + ((size_t)b * NPRIM + n) * (NCLS * DIGD);
    #pragma unroll
    for (int k = 0; k < NCLS; k++) {
      float c = e[k] * inv;
      #pragma unroll
      for (int j = 0; j < DIGD; j++)
        atomicAdd(&ls[k * DIGD + j], c * up[k * DIGD + j]);
    }
  }
  __syncthreads();
  for (int i = threadIdx.x; i < NCLS * DIGD; i += 256)
    atomicAdd(&s[(size_t)b * NCLS * DIGD + i], ls[i]);
}

__global__ void squash_v_kernel(const float* __restrict__ s, float* __restrict__ v,
                                float* __restrict__ vout) {
  int idx = blockIdx.x * blockDim.x + threadIdx.x;
  if (idx >= BATCH * NCLS) return;
  const float* p = s + (size_t)idx * DIGD;
  float sq = 0.f;
  #pragma unroll
  for (int j = 0; j < DIGD; j++) sq += p[j] * p[j];
  float scale = sq / (sqrtf(sq) * (1.f + sq));
  #pragma unroll
  for (int j = 0; j < DIGD; j++) {
    float val = p[j] * scale;
    v[(size_t)idx * DIGD + j] = val;
    vout[(size_t)idx * DIGD + j] = val;   // v portion of d_out
  }
}

__global__ void logit_update(const float* __restrict__ uhat, const float* __restrict__ v,
                             float* __restrict__ blog) {
  long idx = (long)blockIdx.x * blockDim.x + threadIdx.x;
  const long total = (long)BATCH * NPRIM * NCLS;
  if (idx >= total) return;
  int k = (int)(idx % NCLS); long t = idx / NCLS;
  int n = (int)(t % NPRIM); int b = (int)(t / NPRIM);
  const float* up = uhat + (((size_t)b * NPRIM + n) * NCLS + k) * DIGD;
  const float* vp = v + ((size_t)b * NCLS + k) * DIGD;
  float acc = 0.f;
  #pragma unroll
  for (int j = 0; j < DIGD; j++) acc += up[j] * vp[j];
  blog[idx] += acc;
}

// ---------------- decoder ----------------
__global__ void mask_kernel(const float* __restrict__ target, const float* __restrict__ v,
                            float* __restrict__ vm) {
  int idx = blockIdx.x * blockDim.x + threadIdx.x;
  if (idx >= BATCH * DIGD) return;
  int j = idx & 15, b = idx >> 4;
  float acc = 0.f;
  #pragma unroll
  for (int k = 0; k < NCLS; k++)
    acc += target[b * NCLS + k] * v[((size_t)b * NCLS + k) * DIGD + j];
  vm[idx] = acc;
}

__global__ void fc_kernel(const float* __restrict__ in, const float* __restrict__ w,
                          const float* __restrict__ bias, float* __restrict__ out,
                          int IN, int OUT, int act) {
  int idx = blockIdx.x * blockDim.x + threadIdx.x;
  if (idx >= BATCH * OUT) return;
  int o = idx % OUT; int b = idx / OUT;
  const float* ip = in + (size_t)b * IN;
  const float* wp = w + (size_t)o * IN;
  float acc = bias[o];
  for (int i = 0; i < IN; i++) acc += ip[i] * wp[i];
  if (act == 1) acc = fmaxf(acc, 0.f);
  else if (act == 2) acc = 1.f / (1.f + __expf(-acc));
  out[idx] = acc;
}

// ---------------- bilinear 2x upsample, align_corners (36 -> 72) ----------------
__global__ void upsample_kernel(const float* __restrict__ rec, float* __restrict__ out) {
  int idx = blockIdx.x * blockDim.x + threadIdx.x;
  const int total = BATCH * IMG * IMG;
  if (idx >= total) return;
  int x = idx % IMG; int t = idx / IMG;
  int y = t % IMG; int b = t / IMG;
  float sy = (float)y * 35.f / 71.f;
  float sx = (float)x * 35.f / 71.f;
  int y0 = (int)floorf(sy), x0 = (int)floorf(sx);
  int y1 = min(y0 + 1, 35), x1 = min(x0 + 1, 35);
  float wy = sy - (float)y0, wx = sx - (float)x0;
  const float* r = rec + (size_t)b * 1296;
  float v00 = r[y0 * 36 + x0], v01 = r[y0 * 36 + x1];
  float v10 = r[y1 * 36 + x0], v11 = r[y1 * 36 + x1];
  float top = v00 * (1.f - wx) + v01 * wx;
  float bot = v10 * (1.f - wx) + v11 * wx;
  out[idx] = top * (1.f - wy) + bot * wy;
}

// ---------------- host launcher ----------------
extern "C" void kernel_launch(void* const* d_in, const int* in_sizes, int n_in,
                              void* d_out, int out_size, void* d_ws, size_t ws_size,
                              hipStream_t stream) {
  (void)in_sizes; (void)n_in; (void)out_size; (void)ws_size;
  const float* x       = (const float*)d_in[0];
  const float* target  = (const float*)d_in[1];
  const float* conv1_w = (const float*)d_in[2];
  const float* conv1_b = (const float*)d_in[3];
  const float* pc1_w   = (const float*)d_in[4];
  const float* pc1_b   = (const float*)d_in[5];
  const float* pc2_w   = (const float*)d_in[6];
  const float* pc2_b   = (const float*)d_in[7];
  const float* pc3_w   = (const float*)d_in[8];
  const float* pc3_b   = (const float*)d_in[9];
  const float* W_digit = (const float*)d_in[10];
  const float* fc1_w   = (const float*)d_in[11];
  const float* fc1_b   = (const float*)d_in[12];
  const float* fc2_w   = (const float*)d_in[13];
  const float* fc2_b   = (const float*)d_in[14];
  const float* fc3_w   = (const float*)d_in[15];
  const float* fc3_b   = (const float*)d_in[16];

  char* ws = (char*)d_ws;
  float*     h    = (float*)(ws + H_OFF);
  _Float16*  A    = (_Float16*)(ws + A_OFF);
  _Float16*  Wt   = (_Float16*)(ws + WT_OFF);
  float*     pc1o = (float*)(ws + PC1O_OFF);
  float*     pc2o = (float*)(ws + PC2O_OFF);
  float*     u    = (float*)(ws + U_OFF);
  float*     uhat = (float*)(ws + UHAT_OFF);
  float*     blog = (float*)(ws + BLOG_OFF);
  float*     s    = (float*)(ws + S_OFF);
  float*     v    = (float*)(ws + V_OFF);
  float*     vm   = (float*)(ws + VM_OFF);
  float*     h1   = (float*)(ws + H1_OFF);
  float*     h2   = (float*)(ws + H2_OFF);
  float*     rech = (float*)(ws + RECH_OFF);
  float*     out  = (float*)d_out;               // [0,1280) = v ; [1280, ...) = rec

  // 1. conv1 + relu
  {
    long total = (long)BATCH * C1 * FEAT * FEAT;
    conv1_kernel<<<(unsigned)((total + 255) / 256), 256, 0, stream>>>(x, conv1_w, conv1_b, h);
  }
  // 2. im2col (f16) + weight swizzle (f16, [K/32][N][32])
  {
    long total = (long)PC1_M * PC1_K;
    im2col_pc1<<<(unsigned)((total + 255) / 256), 256, 0, stream>>>(h, A);
    long tw = (long)PC1_K * PC1_N;
    wt_pc1<<<(unsigned)((tw + 255) / 256), 256, 0, stream>>>(pc1_w, Wt);
  }
  // 3. pc1 GEMM via WMMA: [7200,6400] x [6400,512] + bias
  {
    dim3 grid((PC1_M + BM - 1) / BM, PC1_N / BN);
    wmma_gemm_bias<<<grid, 256, 0, stream>>>(A, Wt, pc1_b, pc1o, PC1_M, PC1_N, PC1_K);
  }
  // 4. pc2, pc3 (grouped convs)
  {
    long t2 = (long)BATCH * R1 * R1 * 256;
    pc2_kernel<<<(unsigned)((t2 + 255) / 256), 256, 0, stream>>>(pc1o, pc2_w, pc2_b, pc2o);
    long t3 = (long)BATCH * R2 * R2 * 256;
    pc3_kernel<<<(unsigned)((t3 + 255) / 256), 256, 0, stream>>>(pc2o, pc3_w, pc3_b, u);
  }
  // 5. squash primary capsules
  {
    long t = (long)BATCH * NPRIM;
    squash_u_kernel<<<(unsigned)((t + 255) / 256), 256, 0, stream>>>(u);
  }
  // 6. u_hat
  {
    long t = (long)BATCH * NPRIM * NCLS * DIGD;
    uhat_kernel<<<(unsigned)((t + 255) / 256), 256, 0, stream>>>(u, W_digit, uhat);
  }
  // 7. dynamic routing (3 iterations)
  {
    long nb = (long)BATCH * NPRIM * NCLS;
    zero_kernel<<<(unsigned)((nb + 255) / 256), 256, 0, stream>>>(blog, nb);
    for (int it = 0; it < 3; ++it) {
      zero_kernel<<<(1280 + 255) / 256, 256, 0, stream>>>(s, 1280);
      routing_reduce<<<dim3(32, 8), 256, 0, stream>>>(uhat, blog, s);
      squash_v_kernel<<<1, 128, 0, stream>>>(s, v, out);
      if (it < 2)
        logit_update<<<(unsigned)((nb + 255) / 256), 256, 0, stream>>>(uhat, v, blog);
    }
  }
  // 8. decoder
  mask_kernel<<<1, 128, 0, stream>>>(target, v, vm);
  fc_kernel<<<(BATCH * 512 + 255) / 256, 256, 0, stream>>>(vm, fc1_w, fc1_b, h1, 16, 512, 1);
  fc_kernel<<<(BATCH * 1024 + 255) / 256, 256, 0, stream>>>(h1, fc2_w, fc2_b, h2, 512, 1024, 1);
  fc_kernel<<<(BATCH * 1296 + 255) / 256, 256, 0, stream>>>(h2, fc3_w, fc3_b, rech, 1024, 1296, 2);
  // 9. bilinear upsample 36 -> 72 into d_out after v
  {
    long t = (long)BATCH * IMG * IMG;
    upsample_kernel<<<(unsigned)((t + 255) / 256), 256, 0, stream>>>(rech, out + BATCH * NCLS * DIGD);
  }
}
